// FP8Linear_28424093565283
// MI455X (gfx1250) — compile-verified, matrix operand order checked by reading, not run on hardware
//
#include <hip/hip_runtime.h>

// ---------------------------------------------------------------------------
// out = x @ (W_int8 * s)^T  ==  s * (x @ W^T)
// x: [8192,4096] fp32, W: [4096,4096] int8-range (stored int32), s scalar.
// f16 WMMA (int8 weights exact in f16, x ~2^-11 rel err), fp32 accumulation,
// scalar scale in fp32 epilogue. Double-buffered LDS pipeline, BLK_K=64.
// ---------------------------------------------------------------------------

typedef __attribute__((ext_vector_type(16))) _Float16 v16h;
typedef __attribute__((ext_vector_type(8)))  _Float16 v8h;
typedef __attribute__((ext_vector_type(8)))  float    v8f;
typedef __attribute__((ext_vector_type(4)))  float    v4f;
typedef __attribute__((ext_vector_type(4)))  int      v4i;

constexpr int M_TOT = 8192;   // rows of x / out
constexpr int K_TOT = 4096;   // IN
constexpr int N_TOT = 4096;   // OUT

constexpr int BLK_M = 128;
constexpr int BLK_N = 128;
constexpr int BLK_K = 64;                 // two wmma K-substeps per LDS stage
constexpr int KSUB  = 32;                 // one v_wmma_f32_16x16x32_f16 K-step
constexpr int LDSS  = BLK_K + 8;          // 72 halves = 144 B: conflict-free, 16B-aligned rows
constexpr int NSTAGE = K_TOT / BLK_K;     // 64

// 8 waves: 2 (M) x 4 (N); each wave owns 64x32 = 4x2 tiles of 16x16
constexpr int WM = 4;
constexpr int WN = 2;

__global__ __launch_bounds__(256)
void fp8linear_wmma_f16(const float* __restrict__ X,
                        const int*   __restrict__ WQ,
                        const float* __restrict__ SINV,
                        float*       __restrict__ OUT) {
  // double-buffered tiles: 2 * (128*72 + 128*72) halves = 72 KB of 320 KB WGP LDS
  __shared__ _Float16 As[2][BLK_M * LDSS];
  __shared__ _Float16 Bs[2][BLK_N * LDSS];

  const int tid  = threadIdx.x;
  const int lane = tid & 31;
  const int wave = tid >> 5;      // 0..7
  const int wm   = wave >> 2;     // 0..1 : wave row
  const int wn   = wave & 3;      // 0..3 : wave col

  const int gm0 = blockIdx.y * BLK_M;
  const int gn0 = blockIdx.x * BLK_N;

  // Staging: 256 threads, 2 threads per row, 32 contiguous elements each.
  const int ldRow = tid >> 1;           // 0..127
  const int ldCol = (tid & 1) * 32;     // 0 or 32

  const float* xg = X  + (size_t)(gm0 + ldRow) * K_TOT + ldCol;
  const int*   wg = WQ + (size_t)(gn0 + ldRow) * K_TOT + ldCol;

  v8f acc[WM][WN];
#pragma unroll
  for (int i = 0; i < WM; ++i)
#pragma unroll
    for (int j = 0; j < WN; ++j)
      acc[i][j] = (v8f){0.f, 0.f, 0.f, 0.f, 0.f, 0.f, 0.f, 0.f};

  // Fragment addressing: lane holds 16 halves of row (lane%16),
  // k sub-offset (lane/16)*16 -> 16x32 tile striped over 32 lanes.
  const int fr = lane & 15;
  const int fk = (lane >> 4) * 16;

  // ---- pipeline stage helpers ----
  auto load_tile = [&](int k0, v4f xr[8], v4i wr[8]) {
#pragma unroll
    for (int u = 0; u < 8; ++u) xr[u] = *(const v4f*)(xg + k0 + u * 4);
#pragma unroll
    for (int u = 0; u < 8; ++u) wr[u] = *(const v4i*)(wg + k0 + u * 4);
  };

  auto store_tile = [&](int buf, const v4f xr[8], const v4i wr[8]) {
#pragma unroll
    for (int u = 0; u < 4; ++u) {
      v8h ah, bh;
#pragma unroll
      for (int e = 0; e < 4; ++e) {
        ah[e]     = (_Float16)xr[2 * u][e];
        ah[4 + e] = (_Float16)xr[2 * u + 1][e];
        bh[e]     = (_Float16)(float)wr[2 * u][e];
        bh[4 + e] = (_Float16)(float)wr[2 * u + 1][e];
      }
      *(v8h*)&As[buf][ldRow * LDSS + ldCol + u * 8] = ah;   // ds_store_b128
      *(v8h*)&Bs[buf][ldRow * LDSS + ldCol + u * 8] = bh;
    }
  };

  auto compute_stage = [&](int buf) {
#pragma unroll
    for (int kk = 0; kk < BLK_K; kk += KSUB) {
      v16h af[WM], bf[WN];
#pragma unroll
      for (int i = 0; i < WM; ++i) {
        const v8h* p = (const v8h*)&As[buf][(wm * 64 + i * 16 + fr) * LDSS + kk + fk];
        ((v8h*)&af[i])[0] = p[0];
        ((v8h*)&af[i])[1] = p[1];
      }
#pragma unroll
      for (int j = 0; j < WN; ++j) {
        const v8h* p = (const v8h*)&Bs[buf][(wn * 32 + j * 16 + fr) * LDSS + kk + fk];
        ((v8h*)&bf[j])[0] = p[0];
        ((v8h*)&bf[j])[1] = p[1];
      }
#pragma unroll
      for (int i = 0; i < WM; ++i)
#pragma unroll
        for (int j = 0; j < WN; ++j)
          acc[i][j] = __builtin_amdgcn_wmma_f32_16x16x32_f16(
              false, af[i], false, bf[j], (short)0, acc[i][j], false, false);
    }
  };

  // ---- software pipeline: prologue fills buffer 0 ----
  v4f xr[8];
  v4i wr[8];
  load_tile(0, xr, wr);
  store_tile(0, xr, wr);
  __syncthreads();

  for (int stage = 0; stage < NSTAGE; ++stage) {
    const int  cur  = stage & 1;
    const bool more = (stage + 1) < NSTAGE;

    if (more) {
      // issue next tile's global loads up front: latency hidden by 16 WMMAs below
      load_tile((stage + 1) * BLK_K, xr, wr);
      if (stage + 2 < NSTAGE) {
        __builtin_prefetch(xg + (stage + 2) * BLK_K, 0, 1);   // global_prefetch_b8
        __builtin_prefetch(wg + (stage + 2) * BLK_K, 0, 1);
      }
    }

    compute_stage(cur);                  // reads As/Bs[cur]

    if (more) store_tile(cur ^ 1, xr, wr);   // writes the other buffer: no conflict
    __syncthreads();                     // one barrier per 64-K stage
  }

  // ---- epilogue: scale by s, non-temporal fp32 stores ----
  // C/D layout: VGPR r, lanes 0-15 -> M=r, N=lane; lanes 16-31 -> M=8+r, N=lane-16.
  const float s  = SINV[0];
  const int   cn = gn0 + wn * 32 + (lane & 15);
  const int   cm = gm0 + wm * 64 + (lane >> 4) * 8;
#pragma unroll
  for (int i = 0; i < WM; ++i)
#pragma unroll
    for (int j = 0; j < WN; ++j)
#pragma unroll
      for (int r = 0; r < 8; ++r)
        __builtin_nontemporal_store(
            acc[i][j][r] * s,
            &OUT[(size_t)(cm + i * 16 + r) * N_TOT + (cn + j * 16)]);
}

extern "C" void kernel_launch(void* const* d_in, const int* in_sizes, int n_in,
                              void* d_out, int out_size, void* d_ws, size_t ws_size,
                              hipStream_t stream) {
  const float* x    = (const float*)d_in[0];   // [8192, 4096] fp32
  const int*   wq   = (const int*)d_in[1];     // [4096, 4096] int32 (int8 range)
  const float* sinv = (const float*)d_in[2];   // scalar
  float*       out  = (float*)d_out;           // [8192, 4096] fp32

  dim3 grid(N_TOT / BLK_N, M_TOT / BLK_M);     // (32, 64)
  fp8linear_wmma_f16<<<grid, 256, 0, stream>>>(x, wq, sinv, out);
}
